// SelfAttention_71700184039428
// MI455X (gfx1250) — matrix operand order repaired and checked by reading, not compiled, not run on hardware
//
#include <hip/hip_runtime.h>
#include <hip/hip_bf16.h>

typedef _Float16 v8h  __attribute__((ext_vector_type(8)));
typedef _Float16 v16h __attribute__((ext_vector_type(16)));
typedef float    v8f  __attribute__((ext_vector_type(8)));

#define N_TOK 4096
#define CDIM  256
#define SCALE_Q 0.0901684405554356f   /* (1/sqrt(256)) * log2(e): softmax in exp2 domain */

static __device__ __forceinline__ v16h cat8(v8h lo, v8h hi) {
    v16h r;
#pragma unroll
    for (int i = 0; i < 8; ++i) { r[i] = lo[i]; r[i + 8] = hi[i]; }
    return r;
}

static __device__ __forceinline__ v8f wmma16(v16h a, v16h b, v8f c) {
    return __builtin_amdgcn_wmma_f32_16x16x32_f16(false, a, false, b, (short)0, c, false, false);
}

static __device__ __forceinline__ v8f vzero8() {
    v8f z;
#pragma unroll
    for (int i = 0; i < 8; ++i) z[i] = 0.f;
    return z;
}

static __device__ __forceinline__ float half16_max(float v) {
    v = fmaxf(v, __shfl_xor(v, 1));
    v = fmaxf(v, __shfl_xor(v, 2));
    v = fmaxf(v, __shfl_xor(v, 4));
    v = fmaxf(v, __shfl_xor(v, 8));
    return v;
}
static __device__ __forceinline__ float half16_sum(float v) {
    v += __shfl_xor(v, 1);
    v += __shfl_xor(v, 2);
    v += __shfl_xor(v, 4);
    v += __shfl_xor(v, 8);
    return v;
}

// ---------------- weight f32 -> f16 ----------------
__global__ void k_f32_to_f16(const float* __restrict__ s, _Float16* __restrict__ d, int n) {
    int i = blockIdx.x * blockDim.x + threadIdx.x;
    if (i < n) d[i] = (_Float16)s[i];
}

// ---------------- GroupNorm stats: one block per (b,g) ----------------
__global__ __launch_bounds__(256) void k_gn_stats(const float* __restrict__ x, float* __restrict__ stats) {
    int bg = blockIdx.x;                       // b*8+g
    const float4* p = (const float4*)(x + (size_t)bg * 32 * N_TOK);
    float s = 0.f, q = 0.f;
    for (int i = threadIdx.x; i < 32 * N_TOK / 4; i += 256) {
        float4 v = p[i];
        s += v.x + v.y + v.z + v.w;
        q += v.x * v.x + v.y * v.y + v.z * v.z + v.w * v.w;
    }
    __shared__ float rs[256], rq[256];
    rs[threadIdx.x] = s; rq[threadIdx.x] = q;
    __syncthreads();
    for (int st = 128; st > 0; st >>= 1) {
        if (threadIdx.x < st) { rs[threadIdx.x] += rs[threadIdx.x + st]; rq[threadIdx.x] += rq[threadIdx.x + st]; }
        __syncthreads();
    }
    if (threadIdx.x == 0) {
        float inv_n = 1.f / (32.f * N_TOK);
        float mean = rs[0] * inv_n;
        float var  = rq[0] * inv_n - mean * mean;
        stats[bg * 2 + 0] = mean;
        stats[bg * 2 + 1] = rsqrtf(var + 1e-5f);
    }
}

// ------ GroupNorm apply + transpose NCHW -> [b][n][c] f16 (32c x 64n tiles) ------
__global__ __launch_bounds__(256) void k_gn_apply(const float* __restrict__ x, const float* __restrict__ stats,
                                                  const float* __restrict__ gsc, const float* __restrict__ gbi,
                                                  _Float16* __restrict__ xh) {
    __shared__ _Float16 tile[32][72];
    int b = blockIdx.z, ct = blockIdx.y, nt = blockIdx.x;
    int c0 = ct * 32, n0 = nt * 64;
    float mean = stats[(b * 8 + ct) * 2 + 0];
    float rstd = stats[(b * 8 + ct) * 2 + 1];
    int t = threadIdx.x;
    int n_l = t & 63, c_b = t >> 6;            // c_b in 0..3
#pragma unroll
    for (int r = 0; r < 8; ++r) {
        int c_l = c_b * 8 + r;
        float sc = gsc[c0 + c_l] * rstd;
        float bi = gbi[c0 + c_l] - mean * sc;
        float xv = x[((size_t)(b * CDIM + c0 + c_l)) * N_TOK + n0 + n_l];
        tile[c_l][n_l] = (_Float16)(xv * sc + bi);
    }
    __syncthreads();
    int n2 = t >> 2, cs = (t & 3) * 8;
    v8h w;
#pragma unroll
    for (int j = 0; j < 8; ++j) w[j] = tile[cs + j][n2];
    *(v8h*)(xh + ((size_t)b * N_TOK + n0 + n2) * CDIM + c0 + cs) = w;
}

// ---------------- QKV projection: out = (xh @ W^T + bias) * scale, WMMA f16 ----------------
__global__ __launch_bounds__(128) void k_qkv(const _Float16* __restrict__ xh, const _Float16* __restrict__ wh,
                                             const float* __restrict__ bq, const float* __restrict__ bk,
                                             const float* __restrict__ bv,
                                             _Float16* __restrict__ qo, _Float16* __restrict__ ko,
                                             _Float16* __restrict__ vo) {
    int b = blockIdx.y, z = blockIdx.z;
    const _Float16* w = wh + (size_t)z * CDIM * CDIM;
    const float* bias = (z == 0) ? bq : (z == 1) ? bk : bv;
    _Float16* out = (z == 0) ? qo : (z == 1) ? ko : vo;
    float scale = (z == 0) ? SCALE_Q : 1.0f;   // fold softmax scale (exp2 domain) into Q

    int l = threadIdx.x & 31, h = l >> 4, lm = l & 15;
    int n0 = blockIdx.x * 64 + (threadIdx.x >> 5) * 16;
    const _Float16* arow = xh + ((size_t)b * N_TOK + n0 + lm) * CDIM;
    v16h a[8];
#pragma unroll
    for (int kc = 0; kc < 8; ++kc)
        a[kc] = cat8(*(const v8h*)(arow + kc * 32 + h * 8),
                     *(const v8h*)(arow + kc * 32 + h * 8 + 16));
#pragma unroll
    for (int ct = 0; ct < 16; ++ct) {
        int c = ct * 16 + lm;
        v8f acc = vzero8();                    // lets first WMMA take inline SRC2=0
#pragma unroll
        for (int kc = 0; kc < 8; ++kc) {
            v16h bb = cat8(*(const v8h*)(w + (size_t)c * CDIM + kc * 32 + h * 8), *(const v8h*)(w + (size_t)c * CDIM + kc * 32 + h * 8 + 16));
            acc = wmma16(a[kc], bb, acc);
        }
        float bsc = bias[c] * scale;
#pragma unroll
        for (int i = 0; i < 8; ++i)
            out[((size_t)b * N_TOK + n0 + i + 8 * h) * CDIM + c] = (_Float16)fmaf(acc[i], scale, bsc);
    }
}

// ---------------- Flash attention: 4 waves/WG, 16 q-rows/wave, 32-key blocks ----------------
__global__ __launch_bounds__(128) void k_attn(const _Float16* __restrict__ q, const _Float16* __restrict__ k,
                                              const _Float16* __restrict__ v, _Float16* __restrict__ oa) {
    __shared__ _Float16 vt[256 * 48];          // V^T tile: [c][key], row stride 48 halves
    __shared__ _Float16 psc[4][16 * 32];       // per-wave P bounce
    int b = blockIdx.y;
    int wave = threadIdx.x >> 5;
    int l = threadIdx.x & 31, h = l >> 4, lm = l & 15;
    int n0 = blockIdx.x * 64 + wave * 16;

    const _Float16* qrow = q + ((size_t)b * N_TOK + n0 + lm) * CDIM;
    v16h qa[8];
#pragma unroll
    for (int kc = 0; kc < 8; ++kc)
        qa[kc] = cat8(*(const v8h*)(qrow + kc * 32 + h * 8),
                      *(const v8h*)(qrow + kc * 32 + h * 8 + 16));

    v8f o[16];
#pragma unroll
    for (int ct = 0; ct < 16; ++ct) o[ct] = vzero8();
    float m[8], lsum[8];
#pragma unroll
    for (int i = 0; i < 8; ++i) { m[i] = -1e30f; lsum[i] = 0.f; }

    const _Float16* kbase = k + (size_t)b * N_TOK * CDIM;
    const _Float16* vbase = v + (size_t)b * N_TOK * CDIM;

    // double-buffered V tile through registers: thread covers rows wave+4j, cols l*8..l*8+7
    v8h stage[8];
#pragma unroll
    for (int j = 0; j < 8; ++j)
        stage[j] = *(const v8h*)(vbase + ((size_t)(wave + 4 * j)) * CDIM + (size_t)l * 8);

    for (int kb = 0; kb < N_TOK / 32; ++kb) {
        __syncthreads();                       // vt free (readers of previous tile done)
#pragma unroll
        for (int j = 0; j < 8; ++j) {
            int row = wave + 4 * j;
#pragma unroll
            for (int e = 0; e < 8; ++e)
                vt[(l * 8 + e) * 48 + row] = stage[j][e];
        }
        __syncthreads();

        if (kb + 1 < N_TOK / 32) {
            // issue next V tile loads + prefetch next K tile (128 thr * 128B = 16KB)
            const _Float16* nv = vbase + (size_t)(kb + 1) * 32 * CDIM;
#pragma unroll
            for (int j = 0; j < 8; ++j)
                stage[j] = *(const v8h*)(nv + ((size_t)(wave + 4 * j)) * CDIM + (size_t)l * 8);
            __builtin_prefetch((const char*)(kbase + (size_t)(kb + 1) * 32 * CDIM) + threadIdx.x * 128, 0, 3);
        }

        // S = Q K^T (two 16-key subtiles), logits already in exp2 domain
        v8f s0 = vzero8(), s1 = vzero8();
        const _Float16* krow = kbase + (size_t)(kb * 32) * CDIM;
#pragma unroll
        for (int kc = 0; kc < 8; ++kc) {
            v16h b0 = cat8(*(const v8h*)(krow + (size_t)lm * CDIM + kc * 32 + h * 8), *(const v8h*)(krow + (size_t)lm * CDIM + kc * 32 + h * 8 + 16));
            s0 = wmma16(qa[kc], b0, s0);
        }
#pragma unroll
        for (int kc = 0; kc < 8; ++kc) {
            v16h b1 = cat8(*(const v8h*)(krow + (size_t)(16 + lm) * CDIM + kc * 32 + h * 8), *(const v8h*)(krow + (size_t)(16 + lm) * CDIM + kc * 32 + h * 8 + 16));
            s1 = wmma16(qa[kc], b1, s1);
        }

        // online softmax update (rows i+8h, reduce across 16-lane half), base-2 exp
        float p0[8], p1[8], corr[8];
#pragma unroll
        for (int i = 0; i < 8; ++i) {
            float rm = half16_max(fmaxf(s0[i], s1[i]));
            float mn = fmaxf(m[i], rm);
            corr[i] = exp2f(m[i] - mn);
            p0[i] = exp2f(s0[i] - mn);
            p1[i] = exp2f(s1[i] - mn);
            float rs = half16_sum(p0[i] + p1[i]);
            lsum[i] = lsum[i] * corr[i] + rs;
            m[i] = mn;
        }
#pragma unroll
        for (int ct = 0; ct < 16; ++ct)
#pragma unroll
            for (int i = 0; i < 8; ++i) o[ct][i] *= corr[i];

        // P -> LDS (C layout) -> back in A layout
        _Float16* pw = psc[wave];
#pragma unroll
        for (int i = 0; i < 8; ++i) {
            pw[(i + 8 * h) * 32 + lm]      = (_Float16)p0[i];
            pw[(i + 8 * h) * 32 + 16 + lm] = (_Float16)p1[i];
        }
        asm volatile("s_wait_dscnt 0" ::: "memory");
        v16h pa = cat8(*(const v8h*)(pw + lm * 32 + h * 8),
                       *(const v8h*)(pw + lm * 32 + h * 8 + 16));

        // O += P V
#pragma unroll
        for (int ct = 0; ct < 16; ++ct) {
            v16h vb = cat8(*(const v8h*)(vt + (ct * 16 + lm) * 48 + h * 8), *(const v8h*)(vt + (ct * 16 + lm) * 48 + h * 8 + 16));
            o[ct] = wmma16(pa, vb, o[ct]);
        }
    }

    float inv[8];
#pragma unroll
    for (int i = 0; i < 8; ++i) inv[i] = 1.f / lsum[i];
#pragma unroll
    for (int ct = 0; ct < 16; ++ct)
#pragma unroll
        for (int i = 0; i < 8; ++i)
            oa[((size_t)b * N_TOK + n0 + i + 8 * h) * CDIM + ct * 16 + lm] = (_Float16)(o[ct][i] * inv[i]);
}

// -------- Output projection + residual, transpose back to NCHW via LDS --------
__global__ __launch_bounds__(128) void k_oproj(const _Float16* __restrict__ oa, const _Float16* __restrict__ woh,
                                               const float* __restrict__ bo, const float* __restrict__ x,
                                               float* __restrict__ out) {
    __shared__ float obuf[64][129];
    int b = blockIdx.y;
    int wave = threadIdx.x >> 5;
    int l = threadIdx.x & 31, h = l >> 4, lm = l & 15;
    int n0b = blockIdx.x * 64;
    int n0 = n0b + wave * 16;

    const _Float16* arow = oa + ((size_t)b * N_TOK + n0 + lm) * CDIM;
    v16h a[8];
#pragma unroll
    for (int kc = 0; kc < 8; ++kc)
        a[kc] = cat8(*(const v8h*)(arow + kc * 32 + h * 8),
                     *(const v8h*)(arow + kc * 32 + h * 8 + 16));

    for (int half = 0; half < 2; ++half) {
        v8f acc[8];
#pragma unroll
        for (int t = 0; t < 8; ++t) {
            int c = (half * 8 + t) * 16 + lm;
            acc[t] = vzero8();
#pragma unroll
            for (int kc = 0; kc < 8; ++kc) {
                v16h bb = cat8(*(const v8h*)(woh + (size_t)c * CDIM + kc * 32 + h * 8), *(const v8h*)(woh + (size_t)c * CDIM + kc * 32 + h * 8 + 16));
                acc[t] = wmma16(a[kc], bb, acc[t]);
            }
        }
        __syncthreads();
#pragma unroll
        for (int t = 0; t < 8; ++t) {
            float bvv = bo[(half * 8 + t) * 16 + lm];
#pragma unroll
            for (int i = 0; i < 8; ++i)
                obuf[wave * 16 + i + 8 * h][t * 16 + lm] = acc[t][i] + bvv;
        }
        __syncthreads();
        for (int idx = threadIdx.x; idx < 2048; idx += 128) {
            int c_l = idx >> 4;
            int nq = (idx & 15) * 4;
            int c = half * 128 + c_l;
            size_t go = ((size_t)(b * CDIM + c)) * N_TOK + n0b + nq;
            const float4 xr = *(const float4*)(x + go);
            float4 r;
            r.x = obuf[nq + 0][c_l] + xr.x;
            r.y = obuf[nq + 1][c_l] + xr.y;
            r.z = obuf[nq + 2][c_l] + xr.z;
            r.w = obuf[nq + 3][c_l] + xr.w;
            *(float4*)(out + go) = r;
        }
    }
}

extern "C" void kernel_launch(void* const* d_in, const int* in_sizes, int n_in,
                              void* d_out, int out_size, void* d_ws, size_t ws_size,
                              hipStream_t stream) {
    (void)in_sizes; (void)n_in; (void)out_size; (void)ws_size;
    const float* x   = (const float*)d_in[0];
    const float* gsc = (const float*)d_in[1];
    const float* gbi = (const float*)d_in[2];
    const float* wq  = (const float*)d_in[3];
    const float* bq  = (const float*)d_in[4];
    const float* wk  = (const float*)d_in[5];
    const float* bk  = (const float*)d_in[6];
    const float* wv  = (const float*)d_in[7];
    const float* bv  = (const float*)d_in[8];
    const float* wo  = (const float*)d_in[9];
    const float* bo  = (const float*)d_in[10];
    float* out = (float*)d_out;

    char* ws = (char*)d_ws;
    const size_t WSZ = (size_t)CDIM * CDIM;            // 65536
    const size_t TSZ = (size_t)4 * N_TOK * CDIM;       // 4.19M elements
    float*     stats = (float*)(ws);                   // 256 B
    _Float16*  wh    = (_Float16*)(ws + 256);          // 4 * 128 KB (wq,wk,wv,wo f16)
    size_t off = 256 + 4 * WSZ * sizeof(_Float16);
    _Float16*  xh = (_Float16*)(ws + off); off += TSZ * 2;
    _Float16*  qb = (_Float16*)(ws + off); off += TSZ * 2;
    _Float16*  kb = (_Float16*)(ws + off); off += TSZ * 2;
    _Float16*  vb = (_Float16*)(ws + off); off += TSZ * 2;
    _Float16*  ob = (_Float16*)(ws + off); off += TSZ * 2;

    k_f32_to_f16<<<WSZ / 256, 256, 0, stream>>>(wq, wh + 0 * WSZ, (int)WSZ);
    k_f32_to_f16<<<WSZ / 256, 256, 0, stream>>>(wk, wh + 1 * WSZ, (int)WSZ);
    k_f32_to_f16<<<WSZ / 256, 256, 0, stream>>>(wv, wh + 2 * WSZ, (int)WSZ);
    k_f32_to_f16<<<WSZ / 256, 256, 0, stream>>>(wo, wh + 3 * WSZ, (int)WSZ);

    k_gn_stats<<<32, 256, 0, stream>>>(x, stats);
    k_gn_apply<<<dim3(64, 8, 4), 256, 0, stream>>>(x, stats, gsc, gbi, xh);

    k_qkv<<<dim3(64, 4, 3), 128, 0, stream>>>(xh, wh, bq, bk, bv, qb, kb, vb);
    k_attn<<<dim3(64, 4), 128, 0, stream>>>(qb, kb, vb, ob);
    k_oproj<<<dim3(64, 4), 128, 0, stream>>>(ob, wh + 3 * WSZ, bo, x, out);
}
